// myRNNModel_11751030522335
// MI455X (gfx1250) — compile-verified, matrix operand order checked by reading, not run on hardware
//
#include <hip/hip_runtime.h>
#include <hip/hip_bf16.h>
#include <math.h>

// ---- types for CDNA5 WMMA ----
typedef __attribute__((ext_vector_type(16))) _Float16 v16h;
typedef __attribute__((ext_vector_type(8)))  _Float16 v8h;
typedef __attribute__((ext_vector_type(8)))  float    v8f;

#define VOCAB 10000
#define EMB   64
#define HID   128
#define BB    32
#define TT    256
#define NROWS (BB*TT)   // 8192

// Build a 16-half A/B fragment for v_wmma_f32_16x16x32_f16.
// Per ISA layout (16-bit A 16x32 / B 32x16):
//   elements 0..7  <- K = kbase + 8*half + (0..7)      (contiguous 16B)
//   elements 8..15 <- K = kbase + 16 + 8*half + (0..7) (contiguous 16B)
// where half = lane/16 and the row (M for A, N for B) = lane%16.
__device__ __forceinline__ v16h load_frag(const _Float16* __restrict__ p) {
    v8h lo = *(const v8h*)(p);
    v8h hi = *(const v8h*)(p + 16);
    return __builtin_shufflevector(lo, hi, 0,1,2,3,4,5,6,7,8,9,10,11,12,13,14,15);
}

// Branchless tanh: 1 - 2/(exp2(2*log2e*x)+1). Saturates to +/-1 for large |x|.
__device__ __forceinline__ float tanh_fast(float x) {
    float e = __builtin_amdgcn_exp2f(x * 2.8853900817779268f); // 2/ln(2)
    return 1.0f - 2.0f * __builtin_amdgcn_rcpf(e + 1.0f);
}

// Workgroup barrier that only waits for LDS (DScnt), NOT outstanding global
// stores: the rnn16 history stores are consumed only after kernel end, so the
// serial scan chain must not pay global-store-ack latency at every step.
__device__ __forceinline__ void wg_barrier_lds_only() {
    asm volatile("s_wait_dscnt 0x0\n\t"
                 "s_barrier_signal -1\n\t"
                 "s_barrier_wait -1" ::: "memory");
}

// ---------------------------------------------------------------------------
// K0: WdT[v][h] = (f16) Wd[h][v]  (transpose so K=HID is contiguous per vocab row)
// ---------------------------------------------------------------------------
__global__ void k_wd_transpose(const float* __restrict__ Wd, _Float16* __restrict__ WdT) {
    int idx = blockIdx.x * blockDim.x + threadIdx.x;   // over VOCAB*HID
    if (idx >= VOCAB * HID) return;
    int v = idx >> 7;
    int h = idx & (HID - 1);
    WdT[idx] = (_Float16)Wd[h * VOCAB + v];
}

// ---------------------------------------------------------------------------
// K1: xw[n][h] = sum_e emb[ids[n]][e] * W[e][h] + b[h]   (f32, feeds tanh)
// ---------------------------------------------------------------------------
__global__ void k_embed_xw(const int* __restrict__ ids, const float* __restrict__ emb,
                           const float* __restrict__ W, const float* __restrict__ b,
                           float* __restrict__ xw) {
    int idx = blockIdx.x * blockDim.x + threadIdx.x;   // over NROWS*HID
    if (idx >= NROWS * HID) return;
    int n = idx >> 7;
    int h = idx & (HID - 1);
    const float* __restrict__ e = emb + (size_t)ids[n] * EMB;
    float acc = b[h];
    #pragma unroll
    for (int k = 0; k < EMB; ++k) acc = fmaf(e[k], W[k * HID + h], acc);
    xw[idx] = acc;
}

// ---------------------------------------------------------------------------
// K2: sequential RNN scan on ONE workgroup (16 waves, one WGP).
//     - U^T fragments hoisted into registers (invariant across t)
//     - h double-buffered in LDS -> ONE lds-only barrier per step
//     Per step: 8 ds_load_b128 + 8 global b32 + 4 wmma + branchless tanh.
// ---------------------------------------------------------------------------
__global__ void __launch_bounds__(512) k_rnn_scan(const float* __restrict__ U,
                                                  const float* __restrict__ xw,
                                                  _Float16* __restrict__ rnn16) {
    __shared__ _Float16 Ut[HID * HID];     // Ut[n][k] = U[k][n]  (32 KB)
    __shared__ _Float16 hb0[BB * HID];     // h ping buffer (8 KB)
    __shared__ _Float16 hb1[BB * HID];     // h pong buffer (8 KB)

    const int tid = threadIdx.x;
    for (int i = tid; i < HID * HID; i += 512) {
        int k = i >> 7, n = i & (HID - 1);
        Ut[n * HID + k] = (_Float16)U[i];   // U row-major [k][n]
    }
    for (int i = tid; i < BB * HID; i += 512) hb0[i] = (_Float16)0.0f;
    __syncthreads();

    const int lane = tid & 31;
    const int wid  = tid >> 5;
    const int mt   = wid >> 3;     // 0..1   (batch tile)
    const int nt   = wid & 7;      // 0..7   (hid tile)
    const int half = lane >> 4;
    const int l16  = lane & 15;
    const int ncol = nt * 16 + l16;
    const int mbase = mt * 16 + half * 8;  // first of this lane's 8 C rows

    // Hoist B (U^T) fragments: invariant across all 256 timesteps. 32 VGPRs.
    v16h bfrag[4];
    {
        const _Float16* __restrict__ brow = &Ut[ncol * HID];
        #pragma unroll
        for (int kc4 = 0; kc4 < 4; ++kc4)
            bfrag[kc4] = load_frag(brow + kc4 * 32 + 8 * half);
    }

    const _Float16* arow0 = &hb0[(mt * 16 + l16) * HID];  // A-row ptr in buf0
    const _Float16* arow1 = &hb1[(mt * 16 + l16) * HID];  // A-row ptr in buf1

    // One step: read h from `arow`, write h_{t+1} into `hw`, log history.
    auto step = [&](int t, const _Float16* __restrict__ arow,
                    _Float16* __restrict__ hw) {
        float xv[8];
        #pragma unroll
        for (int r = 0; r < 8; ++r)
            xv[r] = xw[((mbase + r) * TT + t) * HID + ncol];

        v8f c = {};
        #pragma unroll
        for (int kc4 = 0; kc4 < 4; ++kc4) {
            v16h a = load_frag(arow + kc4 * 32 + 8 * half);
            c = __builtin_amdgcn_wmma_f32_16x16x32_f16(
                    false, a, false, bfrag[kc4], (short)0, c, false, false);
        }
        #pragma unroll
        for (int r = 0; r < 8; ++r) {
            float hv = tanh_fast(c[r] + xv[r]);
            _Float16 h16 = (_Float16)hv;
            hw[(mbase + r) * HID + ncol] = h16;                       // next state
            rnn16[(size_t)((mbase + r) * TT + t) * HID + ncol] = h16; // history
        }
        // Writes went to the buffer last READ in the previous step (already fenced by
        // the previous barrier), so a single barrier per step is sufficient.
        wg_barrier_lds_only();
    };

    for (int t = 0; t < TT; t += 2) {
        step(t,     arow0, hb1);   // read buf0, write buf1
        step(t + 1, arow1, hb0);   // read buf1, write buf0
    }
}

// ---------------------------------------------------------------------------
// K3: logits = rnn16[8192][128] @ WdT^T + bd   -> out f32 [8192][10000]
//     Each wave: 4 m-tiles x 1 n-tile, Wd fragment held in registers.
//     Output stream is write-once: non-temporal stores keep rnn16/WdT hot in L2.
// ---------------------------------------------------------------------------
__global__ void __launch_bounds__(256) k_out_proj(const _Float16* __restrict__ A,
                                                  const _Float16* __restrict__ Bt,
                                                  const float* __restrict__ bd,
                                                  float* __restrict__ out) {
    const int lane  = threadIdx.x & 31;
    const int wid   = threadIdx.x >> 5;
    const int ntile = blockIdx.x;                    // 0..624  (VOCAB/16)
    const int mbase = (blockIdx.y * 8 + wid) * 4;    // first of 4 m-tiles
    const int half  = lane >> 4;
    const int l16   = lane & 15;
    const int ncol  = ntile * 16 + l16;

    // B fragment (Wd^T rows for this n-tile): invariant across the 4 m-tiles.
    v16h bfrag[4];
    {
        const _Float16* __restrict__ brow = Bt + (size_t)ncol * HID;
        #pragma unroll
        for (int kc4 = 0; kc4 < 4; ++kc4)
            bfrag[kc4] = load_frag(brow + kc4 * 32 + 8 * half);
    }

    const float bias = bd[ncol];

    #pragma unroll
    for (int mi = 0; mi < 4; ++mi) {
        const int mtile = mbase + mi;
        const _Float16* __restrict__ arow = A + (size_t)(mtile * 16 + l16) * HID;
        v8f c = {};
        #pragma unroll
        for (int kc4 = 0; kc4 < 4; ++kc4) {
            v16h a = load_frag(arow + kc4 * 32 + 8 * half);
            c = __builtin_amdgcn_wmma_f32_16x16x32_f16(
                    false, a, false, bfrag[kc4], (short)0, c, false, false);
        }
        #pragma unroll
        for (int r = 0; r < 8; ++r) {
            int m = mtile * 16 + half * 8 + r;
            // write-once 328 MB stream: non-temporal, coalesced 64B per half-wave
            __builtin_nontemporal_store(c[r] + bias, &out[(size_t)m * VOCAB + ncol]);
        }
    }
}

// ---------------------------------------------------------------------------
extern "C" void kernel_launch(void* const* d_in, const int* in_sizes, int n_in,
                              void* d_out, int out_size, void* d_ws, size_t ws_size,
                              hipStream_t stream) {
    const int*   ids = (const int*)d_in[0];
    const float* emb = (const float*)d_in[1];
    const float* W   = (const float*)d_in[2];
    const float* U   = (const float*)d_in[3];
    const float* b   = (const float*)d_in[4];
    const float* Wd  = (const float*)d_in[5];
    const float* bd  = (const float*)d_in[6];
    float* out = (float*)d_out;

    char* ws = (char*)d_ws;
    float*     xw    = (float*)ws;                                         // 4 MB
    _Float16*  rnn16 = (_Float16*)(ws + (size_t)NROWS * HID * 4);          // 2 MB
    _Float16*  WdT   = (_Float16*)(ws + (size_t)NROWS * HID * 4
                                      + (size_t)NROWS * HID * 2);          // 2.56 MB

    k_wd_transpose<<<(VOCAB * HID) / 256, 256, 0, stream>>>(Wd, WdT);
    k_embed_xw<<<(NROWS * HID) / 256, 256, 0, stream>>>(ids, emb, W, b, xw);
    k_rnn_scan<<<1, 512, 0, stream>>>(U, xw, rnn16);
    // 625 n-tiles; 512 m-tiles / (8 waves * 4 per wave) = 16 blocks in y.
    k_out_proj<<<dim3(VOCAB / 16, NROWS / 16 / 32), 256, 0, stream>>>(rnn16, WdT, bd, out);
}
